// ShapedAttention_60490319397277
// MI455X (gfx1250) — matrix operand order repaired and checked by reading, not compiled, not run on hardware
//
#include <hip/hip_runtime.h>
#include <hip/hip_bf16.h>
#include <stdint.h>

// ---------------------------------------------------------------------------
// Types / helpers
// ---------------------------------------------------------------------------
typedef __bf16 bf16_t;
typedef __bf16 v8bf  __attribute__((ext_vector_type(8)));
typedef __bf16 v16bf __attribute__((ext_vector_type(16)));
typedef float  v8f   __attribute__((ext_vector_type(8)));

static __device__ __forceinline__ bf16_t f2bf(float f) {
  uint32_t u = __builtin_bit_cast(uint32_t, f);
  uint32_t r = u + 0x7FFFu + ((u >> 16) & 1u);   // round-to-nearest-even
  return __builtin_bit_cast(bf16_t, (uint16_t)(r >> 16));
}
static __device__ __forceinline__ float bf2f(bf16_t v) {
  uint32_t u = ((uint32_t)__builtin_bit_cast(uint16_t, v)) << 16;
  return __builtin_bit_cast(float, u);
}
static __device__ __forceinline__ v16bf cat16(v8bf lo, v8bf hi) {
  v16bf r;
#pragma unroll
  for (int i = 0; i < 8; ++i) { r[i] = lo[i]; r[8 + i] = hi[i]; }
  return r;
}

// A fragment: 16x32 bf16, rows m0..m0+15 from row-major [*, ldk], contraction k0..k0+31.
// ISA layout: lane L -> row (L&15); element e -> k = (e<8 ? e : e+8) + (L<16 ? 0 : 8).
static __device__ __forceinline__ v16bf load_a_frag(const bf16_t* __restrict__ A, int lane,
                                                    size_t ldk, int m0, int k0) {
  const bf16_t* p = A + (size_t)(m0 + (lane & 15)) * ldk + (size_t)k0 + ((lane < 16) ? 0 : 8);
  v8bf lo = *(const v8bf*)p;          // k = base+0..7
  v8bf hi = *(const v8bf*)(p + 16);   // k = base+16..23
  return cat16(lo, hi);
}
// B fragment: 32x16 bf16 from pre-transposed Bt[N][ldk] (Bt[n][k] = B[k][n]).
// ISA layout: lane L -> col (L&15); element e -> k = (e<8 ? e : e+8) + (L<16 ? 0 : 8).
static __device__ __forceinline__ v16bf load_bt_frag(const bf16_t* __restrict__ Bt, int lane,
                                                     size_t ldk, int n0, int k0) {
  const bf16_t* p = Bt + (size_t)(n0 + (lane & 15)) * ldk + (size_t)k0 + ((lane < 16) ? 0 : 8);
  v8bf lo = *(const v8bf*)p;
  v8bf hi = *(const v8bf*)(p + 16);
  return cat16(lo, hi);
}
static __device__ __forceinline__ v8f wmma_bf16(v16bf a, v16bf b, v8f c) {
  return __builtin_amdgcn_wmma_f32_16x16x32_bf16(false, a, false, b, (short)0, c, false, false);
}

// ---------------------------------------------------------------------------
// CDNA5 async global->LDS copy (guarded; plain LDS copy fallback)
// Builtin signature (from the toolchain diagnostic): first param is a pointer
// to int __attribute__((vector_size(16))) -- i.e. (global int4*, lds int4*,
// imm offset, imm cpol).
// ---------------------------------------------------------------------------
#if __has_builtin(__builtin_amdgcn_global_load_async_to_lds_b128)
#define ATTN_ASYNC 1
typedef int async_v4i __attribute__((__vector_size__(4 * sizeof(int))));
typedef __attribute__((address_space(1))) async_v4i* as1_v4i_ptr;
typedef __attribute__((address_space(3))) async_v4i* as3_v4i_ptr;
#else
#define ATTN_ASYNC 0
#endif

static __device__ __forceinline__ void cp16_g2l(bf16_t* l, const bf16_t* g) {
#if ATTN_ASYNC
  __builtin_amdgcn_global_load_async_to_lds_b128((as1_v4i_ptr)g, (as3_v4i_ptr)l, 0, 0);
#else
  *(v8bf*)l = *(const v8bf*)g;
#endif
}
static __device__ __forceinline__ void cp_wait_all() {
#if ATTN_ASYNC
#if __has_builtin(__builtin_amdgcn_s_wait_asynccnt)
  __builtin_amdgcn_s_wait_asynccnt(0);
#else
  asm volatile("s_wait_asynccnt 0x0" ::: "memory");
#endif
#endif
}

// ---------------------------------------------------------------------------
// Problem constants
// ---------------------------------------------------------------------------
#define BATCH 2
#define SEQ   2048
#define DIM   1024
#define HEADS 16
#define HD    64
#define MTOT  (BATCH * SEQ)   // 4096

// ---------------------------------------------------------------------------
// fp32 -> bf16 conversion / transposed conversion
// ---------------------------------------------------------------------------
__global__ __launch_bounds__(256) void k_cvt_bf16(bf16_t* __restrict__ dst,
                                                  const float* __restrict__ src, int n) {
  int i = blockIdx.x * blockDim.x + threadIdx.x;
  if (i < n) dst[i] = f2bf(src[i]);
}
// dst[N][K] = src[K][N]^T
__global__ __launch_bounds__(256) void k_transpose_bf16(bf16_t* __restrict__ dst,
                                                        const float* __restrict__ src,
                                                        int K, int N) {
  int i = blockIdx.x * blockDim.x + threadIdx.x;
  if (i < N * K) {
    int nrow = i / K, k = i - nrow * K;
    dst[i] = f2bf(src[(size_t)k * N + nrow]);
  }
}

// ---------------------------------------------------------------------------
// QKV projection GEMM: [4096 x 1024] @ [1024 x 3072] -> scatter into
// Q/K/V [bh][n][hd] and Vt [bh][hd][n] bf16 head layouts.
// One wave computes a 64x64 C tile (16 WMMA / k-step, 2x intensity vs 32x32).
// ---------------------------------------------------------------------------
__global__ __launch_bounds__(256) void k_qkv_gemm(const bf16_t* __restrict__ X,
                                                  const bf16_t* __restrict__ Wt,
                                                  bf16_t* __restrict__ Qd,
                                                  bf16_t* __restrict__ Kd,
                                                  bf16_t* __restrict__ Vd,
                                                  bf16_t* __restrict__ Vtd) {
  const int lane = threadIdx.x & 31;
  const int wave = threadIdx.x >> 5;
  const int widx = blockIdx.x * 8 + wave;
  const int tm = widx & 63;            // 64 m-tiles of 64
  const int tn = widx >> 6;            // 48 n-tiles of 64
  const int m0 = tm * 64, n0 = tn * 64;

  v8f c[4][4] = {};
  for (int k0 = 0; k0 < DIM; k0 += 32) {
    if (k0 + 32 < DIM) {
      __builtin_prefetch(X + (size_t)(m0 + (lane & 31)) * DIM + k0 + 32, 0, 1);
      __builtin_prefetch(Wt + (size_t)(n0 + (lane & 31)) * DIM + k0 + 32, 0, 1);
    }
    v16bf a[4], b[4];
#pragma unroll
    for (int i = 0; i < 4; ++i) a[i] = load_a_frag(X, lane, DIM, m0 + 16 * i, k0);
#pragma unroll
    for (int j = 0; j < 4; ++j) b[j] = load_bt_frag(Wt, lane, DIM, n0 + 16 * j, k0);
#pragma unroll
    for (int i = 0; i < 4; ++i)
#pragma unroll
      for (int j = 0; j < 4; ++j)
        c[i][j] = wmma_bf16(a[i], b[j], c[i][j]);
  }

#pragma unroll
  for (int i = 0; i < 4; ++i)
#pragma unroll
    for (int j = 0; j < 4; ++j) {
      const int col  = n0 + j * 16 + (lane & 15);  // global col in [0, 3072)
      const int mat  = col >> 10;                  // 0=Q 1=K 2=V (uniform per tile)
      const int dc   = col & 1023;
      const int head = dc >> 6, hc = dc & 63;
#pragma unroll
      for (int r = 0; r < 8; ++r) {
        const int mrow = m0 + i * 16 + r + ((lane < 16) ? 0 : 8);
        const int bi = mrow >> 11, nn = mrow & 2047;
        const size_t bh  = (size_t)bi * HEADS + head;
        const size_t idx = (bh * SEQ + nn) * HD + hc;
        const bf16_t val = f2bf(c[i][j][r]);
        if (mat == 0)      Qd[idx] = val;
        else if (mat == 1) Kd[idx] = val;
        else { Vd[idx] = val; Vtd[(bh * HD + hc) * SEQ + nn] = val; }
      }
    }
}

// ---------------------------------------------------------------------------
// Per-head column sum of V (for the -gamma/n term)
// ---------------------------------------------------------------------------
__global__ __launch_bounds__(64) void k_vsum(const bf16_t* __restrict__ Vd,
                                             float* __restrict__ Vsum) {
  const int bh = blockIdx.x;      // 32
  const int c  = threadIdx.x;     // 64
  const bf16_t* p = Vd + (size_t)bh * SEQ * HD + c;
  float s = 0.f;
  for (int nn = 0; nn < SEQ; ++nn) s += bf2f(p[(size_t)nn * HD]);
  Vsum[bh * HD + c] = s;
}

// ---------------------------------------------------------------------------
// Flash attention with shaped-attn epilogue.
// Block = 256 threads = 8 waves; each wave owns a 16-row query tile.
// All 8 waves share the same (b,h) and the same key block -> K/Vt tiles are
// staged once per block into LDS via async global->LDS (double buffered),
// giving 8x reuse of each staged byte.
// grid = (b*h) * (SEQ/128) = 512 blocks.
// ---------------------------------------------------------------------------
// Stage K tile (32 keys x 64 hd, contiguous 4KB in global) and Vt tile
// (64 hd x 32 keys, 64B chunks strided by SEQ) into LDS. 256 chunks of 16B
// each per tile; 256 threads -> one chunk per thread per tile.
static __device__ __forceinline__ void stage_kv(bf16_t* __restrict__ lk,
                                                bf16_t* __restrict__ lv,
                                                const bf16_t* __restrict__ Kh,
                                                const bf16_t* __restrict__ Vth,
                                                int j, int tid) {
  // K: flat contiguous copy
  cp16_g2l(lk + tid * 8, Kh + (size_t)j * HD + tid * 8);
  // Vt: row = chunk>>2 (hd), 16B sub-chunk within the 64B row segment
  const int row = tid >> 2, off = (tid & 3) * 8;
  cp16_g2l(lv + row * 32 + off, Vth + (size_t)row * SEQ + j + off);
}

__global__ __launch_bounds__(256) void k_attn(const bf16_t* __restrict__ Qd,
                                              const bf16_t* __restrict__ Kd,
                                              const bf16_t* __restrict__ Vd,
                                              const bf16_t* __restrict__ Vtd,
                                              const float* __restrict__ Vsum,
                                              const float* __restrict__ pAlpha,
                                              const float* __restrict__ pBeta,
                                              const float* __restrict__ pGamma,
                                              bf16_t* __restrict__ Aout) {
  __shared__ bf16_t kbuf[2][32 * HD];       // 2 x 4KB
  __shared__ bf16_t vbuf[2][HD * 32];       // 2 x 4KB
  __shared__ bf16_t pstage[8][16 * 32];     // per-wave P-tile staging (C-frag -> A-frag)

  const int tid  = threadIdx.x;
  const int lane = tid & 31;
  const int wave = tid >> 5;
  const int bh = blockIdx.x >> 4;           // 32 (b,h) pairs
  const int qb = blockIdx.x & 15;           // 16 query blocks of 128 rows
  const int m0 = qb * 128 + wave * 16;
  const int bi = bh >> 4, hh = bh & 15;

  const bf16_t* Qh  = Qd  + (size_t)bh * SEQ * HD;
  const bf16_t* Kh  = Kd  + (size_t)bh * SEQ * HD;
  const bf16_t* Vh  = Vd  + (size_t)bh * SEQ * HD;
  const bf16_t* Vth = Vtd + (size_t)bh * HD * SEQ;

  // scale = d^-0.5 = 1/32 (full dim, per the module); fold in log2(e) for exp2.
  const float sscale = 0.03125f * 1.4426950408889634f;

  const v16bf aQ0 = load_a_frag(Qh, lane, HD, m0, 0);    // hd 0..31
  const v16bf aQ1 = load_a_frag(Qh, lane, HD, m0, 32);   // hd 32..63

  v8f o[4] = {};
  float mrun[8], lrun[8];
#pragma unroll
  for (int r = 0; r < 8; ++r) { mrun[r] = -3.0e38f; lrun[r] = 0.f; }

  bf16_t* myP = pstage[wave];

  stage_kv(kbuf[0], vbuf[0], Kh, Vth, 0, tid);
  cp_wait_all();
  __syncthreads();

  for (int jb = 0; jb < SEQ / 32; ++jb) {
    const int j = jb * 32;
    const int cur = jb & 1;
    if (jb + 1 < SEQ / 32)
      stage_kv(kbuf[cur ^ 1], vbuf[cur ^ 1], Kh, Vth, j + 32, tid);

    const bf16_t* lk = kbuf[cur];
    const bf16_t* lv = vbuf[cur];

    // --- S = (Q K^T) * scale for keys j..j+31 (two 16-col tiles), from LDS ---
    v16bf bk00 = load_bt_frag(lk, lane, HD, 0, 0);
    v16bf bk01 = load_bt_frag(lk, lane, HD, 0, 32);
    v16bf bk10 = load_bt_frag(lk, lane, HD, 16, 0);
    v16bf bk11 = load_bt_frag(lk, lane, HD, 16, 32);
    v8f s0 = {}, s1 = {};
    s0 = wmma_bf16(aQ0, bk00, s0);
    s0 = wmma_bf16(aQ1, bk01, s0);
    s1 = wmma_bf16(aQ0, bk10, s1);
    s1 = wmma_bf16(aQ1, bk11, s1);

    // --- online softmax (rows r / r+8 split across lane halves) ---
    float bmax[8];
#pragma unroll
    for (int r = 0; r < 8; ++r) {
      s0[r] *= sscale; s1[r] *= sscale;
      bmax[r] = fmaxf(s0[r], s1[r]);
    }
#pragma unroll
    for (int off = 1; off < 16; off <<= 1)
#pragma unroll
      for (int r = 0; r < 8; ++r)
        bmax[r] = fmaxf(bmax[r], __shfl_xor(bmax[r], off, 32));

    float psum[8];
#pragma unroll
    for (int r = 0; r < 8; ++r) {
      const float mnew = fmaxf(mrun[r], bmax[r]);
      const float corr = exp2f(mrun[r] - mnew);
      const float p0 = exp2f(s0[r] - mnew);
      const float p1 = exp2f(s1[r] - mnew);
      s0[r] = p0; s1[r] = p1;
      psum[r] = p0 + p1;
      lrun[r] *= corr;
      mrun[r] = mnew;
      o[0][r] *= corr; o[1][r] *= corr; o[2][r] *= corr; o[3][r] *= corr;
    }
#pragma unroll
    for (int off = 1; off < 16; off <<= 1)
#pragma unroll
      for (int r = 0; r < 8; ++r)
        psum[r] += __shfl_xor(psum[r], off, 32);
#pragma unroll
    for (int r = 0; r < 8; ++r) lrun[r] += psum[r];

    // --- C-frag P -> LDS [16 q][32 key] bf16 -> A-frag (same-wave DS is in-order) ---
#pragma unroll
    for (int r = 0; r < 8; ++r) {
      const int rr = r + ((lane < 16) ? 0 : 8);
      myP[rr * 32 + (lane & 15)]      = f2bf(s0[r]);
      myP[rr * 32 + 16 + (lane & 15)] = f2bf(s1[r]);
    }
    asm volatile("" ::: "memory");  // keep compiler from reordering DS load over stores
    {
      const bf16_t* p = myP + (lane & 15) * 32 + ((lane < 16) ? 0 : 8);
      v8bf lo = *(const v8bf*)p;
      v8bf hi = *(const v8bf*)(p + 16);
      const v16bf aP = cat16(lo, hi);
#pragma unroll
      for (int t = 0; t < 4; ++t) {
        v16bf bv = load_bt_frag(lv, lane, 32, t * 16, 0);   // rows=hd, cols=keys (LDS)
        o[t] = wmma_bf16(aP, bv, o[t]);
      }
    }

    cp_wait_all();    // our async issues for the next buffer have landed
    __syncthreads();  // every wave done reading cur + done staging next
  }

  // --- shaped epilogue: alpha*v_q + beta*softmax@V - (gamma/n)*colsum(V) ---
  const float alpha  = pAlpha[0];
  const float beta   = pBeta[0];
  const float gamman = pGamma[0] * (1.0f / (float)SEQ);
  const float* vs = Vsum + bh * HD;
#pragma unroll
  for (int t = 0; t < 4; ++t) {
    const int hc = t * 16 + (lane & 15);
    const float vsum_c = vs[hc];
#pragma unroll
    for (int r = 0; r < 8; ++r) {
      const int q = m0 + r + ((lane < 16) ? 0 : 8);
      const float vq  = bf2f(Vh[(size_t)q * HD + hc]);
      const float res = alpha * vq + beta * (o[t][r] / lrun[r]) - gamman * vsum_c;
      Aout[((size_t)bi * SEQ + q) * DIM + hh * HD + hc] = f2bf(res);
    }
  }
}

// ---------------------------------------------------------------------------
// Output projection: [4096 x 1024] @ [1024 x 1024] + bias -> fp32 d_out
// One wave computes a 64x64 C tile.
// ---------------------------------------------------------------------------
__global__ __launch_bounds__(256) void k_out_gemm(const bf16_t* __restrict__ Ain,
                                                  const bf16_t* __restrict__ Wt,
                                                  const float* __restrict__ bias,
                                                  float* __restrict__ out) {
  const int lane = threadIdx.x & 31;
  const int wave = threadIdx.x >> 5;
  const int widx = blockIdx.x * 8 + wave;
  const int tm = widx & 63;            // 64 m-tiles
  const int tn = widx >> 6;            // 16 n-tiles
  const int m0 = tm * 64, n0 = tn * 64;

  v8f c[4][4] = {};
  for (int k0 = 0; k0 < DIM; k0 += 32) {
    if (k0 + 32 < DIM) {
      __builtin_prefetch(Ain + (size_t)(m0 + (lane & 31)) * DIM + k0 + 32, 0, 1);
      __builtin_prefetch(Wt + (size_t)(n0 + (lane & 31)) * DIM + k0 + 32, 0, 1);
    }
    v16bf a[4], b[4];
#pragma unroll
    for (int i = 0; i < 4; ++i) a[i] = load_a_frag(Ain, lane, DIM, m0 + 16 * i, k0);
#pragma unroll
    for (int j = 0; j < 4; ++j) b[j] = load_bt_frag(Wt, lane, DIM, n0 + 16 * j, k0);
#pragma unroll
    for (int i = 0; i < 4; ++i)
#pragma unroll
      for (int j = 0; j < 4; ++j)
        c[i][j] = wmma_bf16(a[i], b[j], c[i][j]);
  }

#pragma unroll
  for (int i = 0; i < 4; ++i)
#pragma unroll
    for (int j = 0; j < 4; ++j) {
      const int col = n0 + j * 16 + (lane & 15);
      const float bv = bias[col];
#pragma unroll
      for (int r = 0; r < 8; ++r) {
        const int mrow = m0 + i * 16 + r + ((lane < 16) ? 0 : 8);
        out[(size_t)mrow * DIM + col] = c[i][j][r] + bv;
      }
    }
}

// ---------------------------------------------------------------------------
// Launch
// ---------------------------------------------------------------------------
extern "C" void kernel_launch(void* const* d_in, const int* in_sizes, int n_in,
                              void* d_out, int out_size, void* d_ws, size_t ws_size,
                              hipStream_t stream) {
  (void)in_sizes; (void)n_in; (void)out_size; (void)ws_size;

  const float* x     = (const float*)d_in[0];   // [2,2048,1024]
  const float* Wqkv  = (const float*)d_in[1];   // [1024,3072]
  const float* Wout  = (const float*)d_in[2];   // [1024,1024]
  const float* bout  = (const float*)d_in[3];   // [1024]
  const float* alpha = (const float*)d_in[4];
  const float* beta  = (const float*)d_in[5];
  const float* gamma = (const float*)d_in[6];
  float* out = (float*)d_out;                   // [2,2048,1024] fp32

  uint8_t* ws = (uint8_t*)d_ws;
  size_t off = 0;
  auto alloc = [&](size_t bytes) -> void* {
    void* p = ws + off;
    off = (off + bytes + 255) & ~(size_t)255;
    return p;
  };
  bf16_t* xb    = (bf16_t*)alloc((size_t)MTOT * DIM * 2);        // 8 MiB
  bf16_t* wqkvt = (bf16_t*)alloc((size_t)3 * DIM * DIM * 2);     // 6 MiB [3072][1024]
  bf16_t* woutt = (bf16_t*)alloc((size_t)DIM * DIM * 2);         // 2 MiB [1024][1024]
  bf16_t* Qb    = (bf16_t*)alloc((size_t)32 * SEQ * HD * 2);     // 8 MiB
  bf16_t* Kb    = (bf16_t*)alloc((size_t)32 * SEQ * HD * 2);     // 8 MiB
  bf16_t* Vb    = (bf16_t*)alloc((size_t)32 * SEQ * HD * 2);     // 8 MiB
  bf16_t* Vtb   = (bf16_t*)alloc((size_t)32 * HD * SEQ * 2);     // 8 MiB
  bf16_t* attnb = (bf16_t*)alloc((size_t)MTOT * DIM * 2);        // 8 MiB
  float*  vsum  = (float*)alloc((size_t)32 * HD * 4);            // 8 KiB

  const int nX = MTOT * DIM;           // 4194304
  const int nWq = 3 * DIM * DIM;       // 3145728
  const int nWo = DIM * DIM;           // 1048576

  k_cvt_bf16<<<(nX + 255) / 256, 256, 0, stream>>>(xb, x, nX);
  k_transpose_bf16<<<(nWq + 255) / 256, 256, 0, stream>>>(wqkvt, Wqkv, DIM, 3 * DIM);
  k_transpose_bf16<<<(nWo + 255) / 256, 256, 0, stream>>>(woutt, Wout, DIM, DIM);

  // 4096/64 * 3072/64 = 3072 wave-tiles / 8 waves = 384 blocks
  k_qkv_gemm<<<384, 256, 0, stream>>>(xb, wqkvt, Qb, Kb, Vb, Vtb);

  k_vsum<<<32, 64, 0, stream>>>(Vb, vsum);

  // 32 (b,h) * 16 query-blocks = 512 blocks of 8 waves
  k_attn<<<512, 256, 0, stream>>>(Qb, Kb, Vb, Vtb, vsum, alpha, beta, gamma, attnb);

  // 4096/64 * 1024/64 = 1024 wave-tiles / 8 waves = 128 blocks
  k_out_gemm<<<128, 256, 0, stream>>>(attnb, woutt, bout, out);
}
